// Critic_84456236908768
// MI455X (gfx1250) — matrix operand order repaired and checked by reading, not compiled, not run on hardware
//
#include <hip/hip_runtime.h>
#include <math.h>

// ---------------------------------------------------------------------------
// Critic GNN (EdgeConv x2, two q-heads) for MI455X / gfx1250, wave32 + WMMA.
//
// EdgeConv layer-1 is hoisted to node level:
//   W1*[xi | xj-xi] = (W1a-W1b)*xi + W1b*xj  =>  U[dst] + V[src]
// so the per-edge work is only tanh(U+V+b1) and a 128x128 WMMA matvec,
// scattered with global_atomic_max_u32 (monotonic float encoding) to
// implement segment_max. W2/b1/b2 are staged in LDS (320KB/WGP) and shared
// by 8 waves that grid-stride over edge tiles.
//
// Input order (harness): obj_x, actions, tar_scores, geo, wall, category,
// batch, edge_index, then params flattened jax-pytree style (dict keys
// sorted alphabetically at each level): q1 leaves then q2 leaves.
// ---------------------------------------------------------------------------

#define NODES  20000
#define EDGES  320000
#define GRAPHS 32
#define EW     8          // waves per block in the edge kernel

typedef __attribute__((ext_vector_type(16))) _Float16 v16h;
typedef __attribute__((ext_vector_type(8)))  _Float16 v8h;
typedef __attribute__((ext_vector_type(8)))  float    v8f;
typedef _Float16 h16;

// monotonic float<->uint encoding so unsigned atomicMax == float max
static __device__ __forceinline__ unsigned enc_max(float f) {
  unsigned u = __float_as_uint(f);
  return (u & 0x80000000u) ? ~u : (u | 0x80000000u);
}
static __device__ __forceinline__ float dec_max(unsigned u) {
  return (u & 0x80000000u) ? __uint_as_float(u & 0x7fffffffu)
                           : __uint_as_float(~u);
}

// exact-identity tanh via the HW transcendental (v_exp_f32), ~6 VALU ops.
// tanh(x) = sign(x) * (1 - 2/(exp(2|x|)+1)); large |x| -> exp=inf -> 1.
static __device__ __forceinline__ float fast_tanh(float x) {
  float a = fabsf(x);
  float e = __expf(2.0f * a);
  float t = 1.0f - 2.0f / (e + 1.0f);
  return copysignf(t, x);
}

// ---------------------------------------------------------------------------
// Generic WMMA GEMM:  Y[M,OUT] = act( X[M,K] @ W[OUT,K]^T + bias )
// One wave computes a 16x16 tile. K must be a multiple of 32 (padded bufs).
// Optional f32 and/or f16 outputs with independent leading dims.
// ---------------------------------------------------------------------------
__global__ void __launch_bounds__(256)
gemm_wmma(const h16* __restrict__ X, int ldx,
          const h16* __restrict__ W, int ldw,
          const float* __restrict__ bias,
          float* __restrict__ outF, int ldf,
          h16* __restrict__ outH, int ldh,
          int M, int K, int OUT, int act)
{
  const int ntiles = OUT >> 4;
  const int mtiles = (M + 15) >> 4;
  const int wave   = (int)((blockIdx.x * blockDim.x + threadIdx.x) >> 5);
  const int lane   = (int)(threadIdx.x & 31);
  if (wave >= mtiles * ntiles) return;          // wave-uniform exit (EXEC all-1 at WMMA)
  const int mt = wave / ntiles;
  const int nt = wave - mt * ntiles;
  const int hl = lane >> 4;                     // half-wave select
  const int l  = lane & 15;

  int row = mt * 16 + l; if (row >= M) row = M - 1;   // clamp; extra rows never stored
  const h16* xr  = X + (size_t)row * ldx;
  const h16* wr  = W + (size_t)(nt * 16 + l) * ldw + hl * 8;
  const int  akb = hl * 8;

  v8f acc = {};
  for (int k0 = 0; k0 < K; k0 += 32) {
    // A: 16x32 f16, lane(l,hl) holds row m=l, K = {k0+akb..+7, k0+akb+16..+23}
    v8h alo = *(const v8h*)(xr + k0 + akb);
    v8h ahi = *(const v8h*)(xr + k0 + akb + 16);
    // B: 32x16 f16, lane(l,hl) holds col n=l, K = {k0+hl*8..+7, k0+hl*8+16..+23}
    v8h blo = *(const v8h*)(wr + k0);
    v8h bhi = *(const v8h*)(wr + k0 + 16);
    v16h a, b;
#pragma unroll
    for (int c = 0; c < 8; ++c) {
      a[c] = alo[c]; a[c + 8] = ahi[c];
      b[c] = blo[c]; b[c + 8] = bhi[c];
    }
    acc = __builtin_amdgcn_wmma_f32_16x16x32_f16(false, a, false, b,
                                                 (short)0, acc, false, false);
  }

  const int n  = nt * 16 + l;
  const float bv = bias ? bias[n] : 0.0f;
#pragma unroll
  for (int r = 0; r < 8; ++r) {
    const int m = mt * 16 + r + hl * 8;         // C/D layout: vgpr r -> row r (+8 for hi half)
    if (m < M) {
      float v = acc[r] + bv;
      if (act) v = fast_tanh(v);
      if (outF) outF[(size_t)m * ldf + n] = v;
      if (outH) outH[(size_t)m * ldh + n] = (h16)v;
    }
  }
}

// ---------------------------------------------------------------------------
// Fused EdgeConv stage 2. Per block: stage W2[128,128] f16 (+b1,b2) into LDS,
// then 8 waves grid-stride over 16-edge tiles:
//   hidden[e,k] = tanh(U[dst[e],k] + V[src[e],k] + b1[k])   (f16 A-frags)
//   msg = hidden @ W2^T + b2                                (32 WMMAs, B from LDS)
//   segment-max scatter into encoded uint buffer by dst.
// ---------------------------------------------------------------------------
__global__ void __launch_bounds__(32 * EW)
edge_conv_wmma(const int* __restrict__ src,
               const int* __restrict__ dst,
               const float* __restrict__ UV,   // [NODES,256]: U | V
               const float* __restrict__ b1,   // [128]
               const h16*  __restrict__ W2,    // [128,128]
               const float* __restrict__ b2,   // [128]
               unsigned* __restrict__ encOut,  // [NODES,128]
               int Etot, int ntilesTotal)
{
  __shared__ h16   sW2[128 * 128];   // 32 KB
  __shared__ float sb1[128];
  __shared__ float sb2[128];
  __shared__ int   sdst[EW * 16];

  const int tid = (int)threadIdx.x;
  // cooperative stage: 2048 x v8h (16B) chunks, 8 per thread
  for (int i = tid; i < (128 * 128) / 8; i += 32 * EW)
    ((v8h*)sW2)[i] = ((const v8h*)W2)[i];
  if (tid < 128) { sb1[tid] = b1[tid]; sb2[tid] = b2[tid]; }
  __syncthreads();

  const int wv   = tid >> 5;
  const int lane = tid & 31;
  const int hl   = lane >> 4;
  const int l    = lane & 15;
  const int akb  = hl * 8;
  const int stride = (int)gridDim.x * EW;

  for (int tile = (int)blockIdx.x * EW + wv; tile < ntilesTotal; tile += stride) {
    const int e0 = tile * 16;
    int e = e0 + l; if (e >= Etot) e = Etot - 1;
    const int di = dst[e];
    const int si = src[e];
    if (hl == 0) sdst[wv * 16 + l] = (e0 + l < Etot) ? di : -1;
    // wave-private LDS slot; same-wave DS ops are in order (DScnt)

    const float* ud = UV + (size_t)di * 256;         // U row
    const float* vs = UV + (size_t)si * 256 + 128;   // V row

    // Build 4 A-fragments (K = 128) of the tanh-activated hidden layer.
    v16h ha[4];
#pragma unroll
    for (int ks = 0; ks < 4; ++ks) {
      const float* pa = ud + ks * 32 + akb;
      const float* pb = vs + ks * 32 + akb;
      const float* pc = sb1 + ks * 32 + akb;
#pragma unroll
      for (int c = 0; c < 8; ++c) {
        ha[ks][c]     = (h16)fast_tanh(pa[c]      + pb[c]      + pc[c]);
        ha[ks][c + 8] = (h16)fast_tanh(pa[c + 16] + pb[c + 16] + pc[c + 16]);
      }
    }

    // 8 output tiles of 16 channels, 4 WMMAs each (K=128), B-frags from LDS.
    for (int nt = 0; nt < 8; ++nt) {
      const int  n  = nt * 16 + l;
      const h16* wr = sW2 + (size_t)n * 128 + hl * 8;
      v8f acc = {};
#pragma unroll
      for (int ks = 0; ks < 4; ++ks) {
        v8h blo = *(const v8h*)(wr + ks * 32);
        v8h bhi = *(const v8h*)(wr + ks * 32 + 16);
        v16h b;
#pragma unroll
        for (int c = 0; c < 8; ++c) { b[c] = blo[c]; b[c + 8] = bhi[c]; }
        acc = __builtin_amdgcn_wmma_f32_16x16x32_f16(false, ha[ks], false, b,
                                                     (short)0, acc, false, false);
      }
      const float bv = sb2[n];
#pragma unroll
      for (int r = 0; r < 8; ++r) {
        const int m  = r + hl * 8;                // edge row within tile
        const int dn = sdst[wv * 16 + m];
        if (dn >= 0) {
          atomicMax(&encOut[(size_t)dn * 128 + n], enc_max(acc[r] + bv));
        }
      }
    }
  }
}

// ---------------------------------------------------------------------------
// Small helper kernels
// ---------------------------------------------------------------------------
__global__ void pack_pad(const float* __restrict__ w, h16* __restrict__ o,
                         int rows, int cin, int cout)
{
  int i = blockIdx.x * blockDim.x + threadIdx.x;
  if (i >= rows * cout) return;
  int r = i / cout, c = i - r * cout;
  o[i] = (c < cin) ? (h16)w[(size_t)r * cin + c] : (h16)0.0f;
}

// conv.l1.w [128,640] -> [256,320]: rows 0..127 = W1a - W1b ; 128..255 = W1b
__global__ void pack_uv(const float* __restrict__ w, h16* __restrict__ o)
{
  int i = blockIdx.x * blockDim.x + threadIdx.x;
  if (i >= 256 * 320) return;
  int r = i / 320, c = i - r * 320;
  float v = (r < 128) ? (w[(size_t)r * 640 + c] - w[(size_t)r * 640 + 320 + c])
                      : w[(size_t)(r - 128) * 640 + 320 + c];
  o[i] = (h16)v;
}

// wall_feat[g][o] = tanh( W2 . tanh(W1*wall[g]+b1) + b2 )   (32 x 64, tiny)
__global__ void wall_feat_kernel(const float* __restrict__ wall,
                                 const float* __restrict__ w1, const float* __restrict__ b1,
                                 const float* __restrict__ w2, const float* __restrict__ b2,
                                 h16* __restrict__ wf)
{
  __shared__ float h1s[64];
  const int g = (int)blockIdx.x, o = (int)threadIdx.x;
  const float wv = wall[g];
  h1s[o] = fast_tanh(w1[o] * wv + b1[o]);
  __syncthreads();
  float acc = b2[o];
  for (int k = 0; k < 64; ++k) acc += w2[o * 64 + k] * h1s[k];
  wf[g * 64 + o] = (h16)fast_tanh(acc);
}

// per-node prep: pad init-10 & geo inputs, cat-embedding tanh, wall gather
__global__ void prep_nodes(const float* __restrict__ obj_x, const float* __restrict__ actions,
                           const float* __restrict__ tar, const float* __restrict__ geo,
                           const int* __restrict__ category, const int* __restrict__ batch,
                           const float* __restrict__ cat_emb, const h16* __restrict__ wf,
                           h16* __restrict__ in10, h16* __restrict__ gpad,
                           h16* __restrict__ ebuf, h16* __restrict__ x)
{
  int i = blockIdx.x * blockDim.x + threadIdx.x;
  if (i >= NODES) return;
  h16* t = in10 + (size_t)i * 32;
#pragma unroll
  for (int c = 0; c < 4; ++c) t[c]     = (h16)obj_x[(size_t)i * 4 + c];
#pragma unroll
  for (int c = 0; c < 3; ++c) t[4 + c] = (h16)actions[(size_t)i * 3 + c];
#pragma unroll
  for (int c = 0; c < 3; ++c) t[7 + c] = (h16)tar[(size_t)i * 3 + c];
#pragma unroll
  for (int c = 10; c < 32; ++c) t[c] = (h16)0.0f;

  h16* g = gpad + (size_t)i * 32;
  g[0] = (h16)geo[(size_t)i * 2 + 0];
  g[1] = (h16)geo[(size_t)i * 2 + 1];
#pragma unroll
  for (int c = 2; c < 32; ++c) g[c] = (h16)0.0f;

  const int ci = category[i];
  for (int c = 0; c < 64; ++c)
    ebuf[(size_t)i * 64 + c] = (h16)fast_tanh(cat_emb[(size_t)ci * 64 + c]);

  const int bi = batch[i];
  for (int c = 0; c < 64; ++c)
    x[(size_t)i * 320 + 192 + c] = wf[(size_t)bi * 64 + c];
}

__global__ void zero_u32(unsigned* __restrict__ p, int n)
{
  int i = blockIdx.x * blockDim.x + threadIdx.x;
  if (i < n) p[i] = 0u;     // decodes to NaN -> treated as "empty segment"
}

// decode segment-max, where(isfinite,v,0), tanh, write into x[:,0:128]
__global__ void decode_max_kernel(const unsigned* __restrict__ e, h16* __restrict__ x)
{
  int i = blockIdx.x * blockDim.x + threadIdx.x;
  if (i >= NODES * 128) return;
  int m = i >> 7, c = i & 127;
  float v = dec_max(e[i]);
  if (!(v > -INFINITY && v < INFINITY)) v = 0.0f;   // NaN / +-inf -> 0
  x[(size_t)m * 320 + c] = (h16)fast_tanh(v);
}

// final scalar head: q[i] = tail.l2.w . t1[i] + tail.l2.b
__global__ void tail_out(const h16* __restrict__ t1, const float* __restrict__ w,
                         const float* __restrict__ b, float* __restrict__ out)
{
  int i = blockIdx.x * blockDim.x + threadIdx.x;
  if (i >= NODES) return;
  float acc = b[0];
  for (int c = 0; c < 128; ++c) acc += (float)t1[(size_t)i * 128 + c] * w[c];
  out[i] = acc;
}

// ---------------------------------------------------------------------------
// Host orchestration
// ---------------------------------------------------------------------------
// param leaf indices (alphabetical pytree flatten within each q-net dict)
enum {
  P_CAT_EMB = 0, P_CATLIN_B, P_CATLIN_W,
  P_C1L1_B, P_C1L1_W, P_C1L2_B, P_C1L2_W,
  P_C2L1_B, P_C2L1_W, P_C2L2_B, P_C2L2_W,
  P_GEO1_B, P_GEO1_W, P_GEO2_B, P_GEO2_W,
  P_INIT1_B, P_INIT1_W, P_INIT2_B, P_INIT2_W,
  P_TAIL1_B, P_TAIL1_W, P_TAIL2_B, P_TAIL2_W,
  P_WALL1_B, P_WALL1_W, P_WALL2_B, P_WALL2_W,
  NPARAM = 27
};

extern "C" void kernel_launch(void* const* d_in, const int* in_sizes, int n_in,
                              void* d_out, int out_size, void* d_ws, size_t ws_size,
                              hipStream_t stream)
{
  (void)in_sizes; (void)n_in; (void)out_size; (void)ws_size;

  const float* obj_x    = (const float*)d_in[0];
  const float* actions  = (const float*)d_in[1];
  const float* tar      = (const float*)d_in[2];
  const float* geo      = (const float*)d_in[3];
  const float* wall     = (const float*)d_in[4];
  const int*   category = (const int*)d_in[5];
  const int*   batch    = (const int*)d_in[6];
  const int*   edge     = (const int*)d_in[7];
  const int*   srcI     = edge;            // edge_index[0]
  const int*   dstI     = edge + EDGES;    // edge_index[1]

  auto P = [&](int q, int idx) -> const float* {
    return (const float*)d_in[8 + q * NPARAM + idx];
  };

  // bump-allocate workspace (~62 MB), reused across q1/q2
  char* wsp = (char*)d_ws;
  auto take = [&](size_t bytes) -> char* {
    char* r = wsp; wsp += (bytes + 255) & ~(size_t)255; return r;
  };
  h16*      xbuf  = (h16*)take((size_t)NODES * 320 * 2);   // [h|class|wall|geo]
  float*    UV    = (float*)take((size_t)NODES * 256 * 4); // U | V
  unsigned* encb  = (unsigned*)take((size_t)NODES * 128 * 4);
  h16*      in10  = (h16*)take((size_t)NODES * 32 * 2);
  h16*      h1b   = (h16*)take((size_t)NODES * 128 * 2);
  h16*      ebuf  = (h16*)take((size_t)NODES * 64 * 2);
  h16*      gpad  = (h16*)take((size_t)NODES * 32 * 2);
  h16*      g1b   = (h16*)take((size_t)NODES * 64 * 2);
  h16*      t1b   = (h16*)take((size_t)NODES * 128 * 2);
  h16*      wfb   = (h16*)take((size_t)GRAPHS * 64 * 2);
  h16*      wInit1 = (h16*)take(128 * 32 * 2);
  h16*      wInit2 = (h16*)take(128 * 128 * 2);
  h16*      wCat   = (h16*)take(64 * 64 * 2);
  h16*      wGeo1  = (h16*)take(64 * 32 * 2);
  h16*      wGeo2  = (h16*)take(64 * 64 * 2);
  h16*      wUVw   = (h16*)take(256 * 320 * 2);
  h16*      wL2w   = (h16*)take(128 * 128 * 2);
  h16*      wTailw = (h16*)take(128 * 320 * 2);

  auto gemm = [&](const h16* X, int ldx, const h16* W, int ldw, const float* bias,
                  float* oF, int ldf, h16* oH, int ldh, int M, int K, int OUT, int act) {
    int waves  = ((M + 15) / 16) * (OUT / 16);
    int blocks = (waves + 7) / 8;
    gemm_wmma<<<blocks, 256, 0, stream>>>(X, ldx, W, ldw, bias, oF, ldf, oH, ldh,
                                          M, K, OUT, act);
  };
  auto pack = [&](const float* w, h16* o, int rows, int cin, int cout) {
    int n = rows * cout;
    pack_pad<<<(n + 255) / 256, 256, 0, stream>>>(w, o, rows, cin, cout);
  };

  float* out = (float*)d_out;

  for (int q = 0; q < 2; ++q) {
    // ---- weight packing (f16) ----
    pack(P(q, P_INIT1_W), wInit1, 128, 10, 32);
    pack(P(q, P_INIT2_W), wInit2, 128, 128, 128);
    pack(P(q, P_CATLIN_W), wCat, 64, 64, 64);
    pack(P(q, P_GEO1_W), wGeo1, 64, 2, 32);
    pack(P(q, P_GEO2_W), wGeo2, 64, 64, 64);
    pack(P(q, P_TAIL1_W), wTailw, 128, 320, 320);

    // ---- conditional features + init encoder inputs ----
    wall_feat_kernel<<<GRAPHS, 64, 0, stream>>>(wall, P(q, P_WALL1_W), P(q, P_WALL1_B),
                                                P(q, P_WALL2_W), P(q, P_WALL2_B), wfb);
    prep_nodes<<<(NODES + 255) / 256, 256, 0, stream>>>(obj_x, actions, tar, geo,
                                                        category, batch,
                                                        P(q, P_CAT_EMB), wfb,
                                                        in10, gpad, ebuf, xbuf);

    // init_enc: 32->128 (tanh), 128->128 (tanh) -> x[:,0:128]
    gemm(in10, 32, wInit1, 32, P(q, P_INIT1_B), nullptr, 0, h1b, 128, NODES, 32, 128, 1);
    gemm(h1b, 128, wInit2, 128, P(q, P_INIT2_B), nullptr, 0, xbuf, 320, NODES, 128, 128, 1);
    // class feat -> x[:,128:192]
    gemm(ebuf, 64, wCat, 64, P(q, P_CATLIN_B), nullptr, 0, xbuf + 128, 320, NODES, 64, 64, 1);
    // geo feat -> x[:,256:320]
    gemm(gpad, 32, wGeo1, 32, P(q, P_GEO1_B), nullptr, 0, g1b, 64, NODES, 32, 64, 1);
    gemm(g1b, 64, wGeo2, 64, P(q, P_GEO2_B), nullptr, 0, xbuf + 256, 320, NODES, 64, 64, 1);

    // ---- two EdgeConv rounds ----
    for (int conv = 0; conv < 2; ++conv) {
      const float* l1w = P(q, conv ? P_C2L1_W : P_C1L1_W);
      const float* l1c = P(q, conv ? P_C2L1_B : P_C1L1_B);
      const float* l2w = P(q, conv ? P_C2L2_W : P_C1L2_W);
      const float* l2c = P(q, conv ? P_C2L2_B : P_C1L2_B);
      pack_uv<<<(256 * 320 + 255) / 256, 256, 0, stream>>>(l1w, wUVw);
      pack(l2w, wL2w, 128, 128, 128);
      // node-level layer 1: UV[N,256] = x @ [A;B]^T   (no bias, no act)
      gemm(xbuf, 320, wUVw, 320, nullptr, UV, 256, nullptr, 0, NODES, 320, 256, 0);
      zero_u32<<<(NODES * 128 + 255) / 256, 256, 0, stream>>>(encb, NODES * 128);
      const int ntiles = (EDGES + 15) / 16;
      edge_conv_wmma<<<640, 32 * EW, 0, stream>>>(srcI, dstI, UV, l1c, wL2w, l2c,
                                                  encb, EDGES, ntiles);
      decode_max_kernel<<<(NODES * 128 + 255) / 256, 256, 0, stream>>>(encb, xbuf);
    }

    // ---- tail: 320->128 (tanh), 128->1 ----
    gemm(xbuf, 320, wTailw, 320, P(q, P_TAIL1_B), nullptr, 0, t1b, 128, NODES, 320, 128, 1);
    tail_out<<<(NODES + 255) / 256, 256, 0, stream>>>(t1b, P(q, P_TAIL2_W),
                                                      P(q, P_TAIL2_B), out + q * NODES);
  }
}
